// StackGNN_41308995452954
// MI455X (gfx1250) — matrix operand order repaired and checked by reading, not compiled, not run on hardware
//
#include <hip/hip_runtime.h>
#include <hip/hip_bf16.h>

typedef _Float16 half_t;
typedef __attribute__((ext_vector_type(16))) _Float16 v16h;
typedef __attribute__((ext_vector_type(8)))  float    v8f;

#define NN 20000      // nodes
#define NE 320000     // edges
#define HD 128        // hidden dim

enum { MODE_PLAIN = 0, MODE_EDGECAT = 1, MODE_NODECAT = 2 };

// ---------------------------------------------------------------------------
// Weight convert+transpose: W[K][N] f32 (row-major, as in x @ W) -> WT[N][K] f16
// ---------------------------------------------------------------------------
__global__ void cvt_wT_kernel(const float* __restrict__ W, half_t* __restrict__ WT,
                              int K, int N) {
  int idx = blockIdx.x * blockDim.x + threadIdx.x;
  if (idx >= K * N) return;
  int k = idx / N, n = idx - k * N;
  WT[(size_t)n * K + k] = (half_t)W[idx];
}

// ---------------------------------------------------------------------------
// Fused-gather WMMA GEMM:  C[M][128] = act( rows(A) @ W + bias )
//   MODE_PLAIN  : row r of A is A[r*K .. r*K+K)
//   MODE_EDGECAT: row e = [ h_n[src[e]] | h_n[dst[e]] | h_e[e] ]   (K=384)
//   MODE_NODECAT: row v = [ h_n[v] | agg[v] ]                      (K=256)
// Block = 256 threads (8 wave32); wave w owns 2 row tiles (32 rows) and all
// 8 column tiles of N=128 -> block covers 256 rows. Each B fragment (from
// LDS, padded pitch => conflict-free ds_load_b128) feeds 2 WMMAs.
// ---------------------------------------------------------------------------
template <int MODE, int K, bool RELU>
__global__ void __launch_bounds__(256)
__attribute__((amdgpu_waves_per_eu(4)))
gemm_wmma_kernel(const float* __restrict__ A,   // PLAIN:[M][K]; EDGECAT: h_e; NODECAT: agg
                 const float* __restrict__ A2,  // h_n for gather modes
                 const int* __restrict__ src, const int* __restrict__ dst,
                 const half_t* __restrict__ WT, // [128][K] f16 (pre-transposed)
                 const float* __restrict__ bias,
                 float* __restrict__ C, int M) {
  constexpr int KC    = (K < 128) ? K : 128;   // K-chunk staged in LDS
  constexpr int PITCH = KC + 8;                // halfs; pad kills bank conflicts
  __shared__ __align__(16) half_t Bs[128 * PITCH];

  const int lane    = threadIdx.x & 31;
  const int wave    = threadIdx.x >> 5;
  const int rowBase = (blockIdx.x * 8 + wave) * 32;
  const int r       = lane & 15;
  const int kbA = (lane < 16) ? 0 : 16;        // A frag: K {kbA..kbA+15}
  const int kbB = (lane < 16) ? 0 : 16;        // B frag: K {kbB..kbB+15}

  int rowc[2], sIdx[2] = {0, 0}, dIdx[2] = {0, 0};
#pragma unroll
  for (int g = 0; g < 2; g++) {
    int row = rowBase + g * 16 + r;
    rowc[g] = (row < M) ? row : (M - 1);       // clamp for safe loads
    if (MODE == MODE_EDGECAT) { sIdx[g] = src[rowc[g]]; dIdx[g] = dst[rowc[g]]; }
  }

  v8f acc[2][8] = {};

#pragma unroll
  for (int kc = 0; kc < K; kc += KC) {
    // ---- cooperative stage of WT[:, kc:kc+KC] into LDS (16B per thread-iter)
    __syncthreads();
    for (int idx = threadIdx.x; idx < 128 * (KC / 8); idx += 256) {
      int rw = idx / (KC / 8), c8 = idx - rw * (KC / 8);
      *(uint4*)&Bs[rw * PITCH + c8 * 8] =
          *(const uint4*)(WT + (size_t)rw * K + kc + c8 * 8);
    }
    __syncthreads();

#pragma unroll
    for (int kk = 0; kk < KC; kk += 32) {
      const int k0 = kc + kk;
      v16h a[2];
#pragma unroll
      for (int g = 0; g < 2; g++) {
        const float* rp;
        if (MODE == MODE_PLAIN) {
          rp = A + (size_t)rowc[g] * K + k0;
        } else if (MODE == MODE_EDGECAT) {
          if (k0 < 128)      rp = A2 + (size_t)sIdx[g] * HD + k0;
          else if (k0 < 256) rp = A2 + (size_t)dIdx[g] * HD + (k0 - 128);
          else               rp = A  + (size_t)rowc[g] * HD + (k0 - 256);
        } else { // MODE_NODECAT
          if (k0 < 128)      rp = A2 + (size_t)rowc[g] * HD + k0;
          else               rp = A  + (size_t)rowc[g] * HD + (k0 - 128);
        }
        const float4 f0 = *(const float4*)(rp + kbA);
        const float4 f1 = *(const float4*)(rp + kbA + 4);
        const float4 f2 = *(const float4*)(rp + kbA + 8);
        const float4 f3 = *(const float4*)(rp + kbA + 12);
        a[g][0]  = (half_t)f0.x; a[g][1]  = (half_t)f0.y;
        a[g][2]  = (half_t)f0.z; a[g][3]  = (half_t)f0.w;
        a[g][4]  = (half_t)f1.x; a[g][5]  = (half_t)f1.y;
        a[g][6]  = (half_t)f1.z; a[g][7]  = (half_t)f1.w;
        a[g][8]  = (half_t)f2.x; a[g][9]  = (half_t)f2.y;
        a[g][10] = (half_t)f2.z; a[g][11] = (half_t)f2.w;
        a[g][12] = (half_t)f3.x; a[g][13] = (half_t)f3.y;
        a[g][14] = (half_t)f3.z; a[g][15] = (half_t)f3.w;
      }

#pragma unroll
      for (int t = 0; t < 8; t++) {
        const half_t* bp = &Bs[(t * 16 + r) * PITCH + kk + kbB];
        v16h b;
        __builtin_memcpy(&b, bp, 32);   // 2x ds_load_b128
        acc[0][t] = __builtin_amdgcn_wmma_f32_16x16x32_f16(
            false, a[0], false, b, (short)0, acc[0][t], false, false);
        acc[1][t] = __builtin_amdgcn_wmma_f32_16x16x32_f16(
            false, a[1], false, b, (short)0, acc[1][t], false, false);
      }
    }
  }

  // Epilogue: C VGPR j, lane l -> row = j + (l<16?0:8), col = l%16
#pragma unroll
  for (int g = 0; g < 2; g++) {
#pragma unroll
    for (int t = 0; t < 8; t++) {
      const int c  = t * 16 + r;
      const float bv = bias[c];
#pragma unroll
      for (int j = 0; j < 8; j++) {
        const int rr = rowBase + g * 16 + j + ((lane < 16) ? 0 : 8);
        if (rr < M) {
          float v = acc[g][t][j] + bv;
          if (RELU) v = fmaxf(v, 0.0f);
          C[(size_t)rr * HD + c] = v;
        }
      }
    }
  }
}

// ---------------------------------------------------------------------------
// Attention logits: logit[e] = dot(h_e[e], aW) + ab  (one wave per edge)
// ---------------------------------------------------------------------------
__global__ void __launch_bounds__(256)
logit_kernel(const float* __restrict__ he, const float* __restrict__ aW,
             const float* __restrict__ ab, float* __restrict__ logit, int E) {
  int gid  = blockIdx.x * blockDim.x + threadIdx.x;
  int e    = gid >> 5;
  int lane = threadIdx.x & 31;
  if (e >= E) return;
  const float* rp = he + (size_t)e * HD + lane * 4;
  const float* wp = aW + lane * 4;
  float s = rp[0] * wp[0] + rp[1] * wp[1] + rp[2] * wp[2] + rp[3] * wp[3];
#pragma unroll
  for (int off = 16; off > 0; off >>= 1) s += __shfl_down(s, off, 32);
  if (lane == 0) logit[e] = s + ab[0];
}

// ---------------------------------------------------------------------------
// Per-layer init: agg = 0 (N*128), mx = -inf, den = 0 (N)
// ---------------------------------------------------------------------------
__global__ void init_layer_kernel(float* __restrict__ agg, float* __restrict__ mx,
                                  float* __restrict__ den, int N) {
  int i = blockIdx.x * blockDim.x + threadIdx.x;
  if (i < N * HD) agg[i] = 0.0f;
  if (i < N) { mx[i] = -__builtin_inff(); den[i] = 0.0f; }
}

__device__ inline void atomicMaxFloat(float* addr, float val) {
  if (val >= 0.0f)
    atomicMax((int*)addr, __float_as_int(val));
  else
    atomicMin((unsigned int*)addr, __float_as_uint(val));
}

__global__ void attn_max_kernel(const float* __restrict__ logit,
                                const int* __restrict__ dst,
                                float* __restrict__ mx, int E) {
  int e = blockIdx.x * blockDim.x + threadIdx.x;
  if (e >= E) return;
  atomicMaxFloat(&mx[dst[e]], logit[e]);
}

__global__ void attn_exp_kernel(float* __restrict__ w,  // in: logits, out: exp weights
                                const int* __restrict__ dst,
                                const float* __restrict__ mx,
                                float* __restrict__ den, int E) {
  int e = blockIdx.x * blockDim.x + threadIdx.x;
  if (e >= E) return;
  int d = dst[e];
  float v = __expf(w[e] - mx[d]);
  w[e] = v;
  atomicAdd(&den[d], v);
}

// agg[dst[e]][h] += h_e[e][h] * w[e]/(den[dst[e]]+1e-9)
__global__ void agg_scatter_kernel(const float* __restrict__ he,
                                   const float* __restrict__ w,
                                   const float* __restrict__ den,
                                   const int* __restrict__ dst,
                                   float* __restrict__ agg, int E) {
  int gid = blockIdx.x * blockDim.x + threadIdx.x;
  int e = gid >> 7, h = gid & 127;
  if (e >= E) return;
  int d = dst[e];
  float attn = w[e] / (den[d] + 1e-9f);
  atomicAdd(&agg[(size_t)d * HD + h], he[(size_t)e * HD + h] * attn);
}

// ---------------------------------------------------------------------------
// Regression head: 128 -relu-> 64 -relu-> 32 -> 1.  One wave per node.
// ---------------------------------------------------------------------------
__global__ void __launch_bounds__(256)
head_kernel(const float* __restrict__ hn,
            const float* __restrict__ W0, const float* __restrict__ b0,
            const float* __restrict__ W1, const float* __restrict__ b1,
            const float* __restrict__ W2, const float* __restrict__ b2,
            float* __restrict__ out, int N) {
  __shared__ float srow[8][HD];
  __shared__ float sx1[8][64];
  int wave = threadIdx.x >> 5, lane = threadIdx.x & 31;
  int node = blockIdx.x * 8 + wave;
  int nc   = (node < N) ? node : (N - 1);
#pragma unroll
  for (int i = 0; i < 4; i++)
    srow[wave][lane * 4 + i] = hn[(size_t)nc * HD + lane * 4 + i];
  __syncthreads();
#pragma unroll
  for (int jj = 0; jj < 2; jj++) {
    int j = lane + jj * 32;
    float s = b0[j];
    for (int k = 0; k < HD; k++) s += srow[wave][k] * W0[k * 64 + j];
    sx1[wave][j] = fmaxf(s, 0.0f);
  }
  __syncthreads();
  float s2 = b1[lane];
  for (int k = 0; k < 64; k++) s2 += sx1[wave][k] * W1[k * 32 + lane];
  s2 = fmaxf(s2, 0.0f);
  float c = s2 * W2[lane];
#pragma unroll
  for (int off = 16; off > 0; off >>= 1) c += __shfl_down(c, off, 32);
  if (lane == 0 && node < N) out[node] = c + b2[0];
}

// ---------------------------------------------------------------------------
// Orchestration
// ---------------------------------------------------------------------------
extern "C" void kernel_launch(void* const* d_in, const int* in_sizes, int n_in,
                              void* d_out, int out_size, void* d_ws, size_t ws_size,
                              hipStream_t stream) {
  (void)in_sizes; (void)n_in; (void)out_size; (void)ws_size;
  // ---- input pointers (insertion-order DFS flattening of setup_inputs) ----
  const float* nf  = (const float*)d_in[0];   // [NN,128]
  const float* ef  = (const float*)d_in[1];   // [NE,32]
  const int*   src = (const int*)  d_in[2];   // [NE]
  const int*   dst = (const int*)  d_in[3];   // [NE]
  const float* enc_n_W = (const float*)d_in[4];
  const float* enc_n_b = (const float*)d_in[5];
  const float* enc_e_W = (const float*)d_in[6];
  const float* enc_e_b = (const float*)d_in[7];
  const float *eW1[2], *eb1[2], *eW2[2], *eb2[2], *aW[2], *ab[2];
  const float *nW1[2], *nb1[2], *nW2[2], *nb2[2];
  for (int l = 0; l < 2; l++) {
    int base = 8 + l * 10;
    eW1[l] = (const float*)d_in[base + 0];  eb1[l] = (const float*)d_in[base + 1];
    eW2[l] = (const float*)d_in[base + 2];  eb2[l] = (const float*)d_in[base + 3];
    aW [l] = (const float*)d_in[base + 4];  ab [l] = (const float*)d_in[base + 5];
    nW1[l] = (const float*)d_in[base + 6];  nb1[l] = (const float*)d_in[base + 7];
    nW2[l] = (const float*)d_in[base + 8];  nb2[l] = (const float*)d_in[base + 9];
  }
  const float* hW0 = (const float*)d_in[28];
  const float* hW1 = (const float*)d_in[29];
  const float* hW2 = (const float*)d_in[30];
  const float* hb0 = (const float*)d_in[31];
  const float* hb1 = (const float*)d_in[32];
  const float* hb2 = (const float*)d_in[33];
  float* out = (float*)d_out;

  // ---- workspace carve-up ----
  char* p = (char*)d_ws;
  auto carve = [&](size_t bytes) {
    void* r = (void*)p;
    p += (bytes + 255) & ~(size_t)255;
    return r;
  };
  float* h_n   = (float*)carve((size_t)NN * HD * 4);
  float* h_e   = (float*)carve((size_t)NE * HD * 4);
  float* tmp_e = (float*)carve((size_t)NE * HD * 4);
  float* agg   = (float*)carve((size_t)NN * HD * 4);
  float* tmp_n = (float*)carve((size_t)NN * HD * 4);
  float* wbuf  = (float*)carve((size_t)NE * 4);
  float* mx    = (float*)carve((size_t)NN * 4);
  float* den   = (float*)carve((size_t)NN * 4);
  half_t* enc_nT = (half_t*)carve((size_t)128 * 128 * 2);
  half_t* enc_eT = (half_t*)carve((size_t)128 * 32 * 2);
  half_t *eW1T[2], *eW2T[2], *nW1T[2], *nW2T[2];
  for (int l = 0; l < 2; l++) {
    eW1T[l] = (half_t*)carve((size_t)128 * 384 * 2);
    eW2T[l] = (half_t*)carve((size_t)128 * 128 * 2);
    nW1T[l] = (half_t*)carve((size_t)128 * 256 * 2);
    nW2T[l] = (half_t*)carve((size_t)128 * 128 * 2);
  }

  auto cvt = [&](const float* W, half_t* WT, int K, int N) {
    int total = K * N;
    cvt_wT_kernel<<<(total + 255) / 256, 256, 0, stream>>>(W, WT, K, N);
  };
  // ---- prep: f16 transposed weights ----
  cvt(enc_n_W, enc_nT, 128, 128);
  cvt(enc_e_W, enc_eT, 32, 128);
  for (int l = 0; l < 2; l++) {
    cvt(eW1[l], eW1T[l], 384, 128);
    cvt(eW2[l], eW2T[l], 128, 128);
    cvt(nW1[l], nW1T[l], 256, 128);
    cvt(nW2[l], nW2T[l], 128, 128);
  }

  const int gN = (NN + 255) / 256;   // node-GEMM blocks (256 rows/block)
  const int gE = (NE + 255) / 256;   // edge-GEMM blocks

  // ---- encoders ----
  gemm_wmma_kernel<MODE_PLAIN, 128, false><<<gN, 256, 0, stream>>>(
      nf, nullptr, nullptr, nullptr, enc_nT, enc_n_b, h_n, NN);
  gemm_wmma_kernel<MODE_PLAIN, 32, false><<<gE, 256, 0, stream>>>(
      ef, nullptr, nullptr, nullptr, enc_eT, enc_e_b, h_e, NE);

  // ---- GN layers ----
  for (int l = 0; l < 2; l++) {
    // edge MLP: hidden = relu([h_src|h_dst|h_e] @ eW1 + eb1); h_e = hidden @ eW2 + eb2
    gemm_wmma_kernel<MODE_EDGECAT, 384, true><<<gE, 256, 0, stream>>>(
        h_e, h_n, src, dst, eW1T[l], eb1[l], tmp_e, NE);
    gemm_wmma_kernel<MODE_PLAIN, 128, false><<<gE, 256, 0, stream>>>(
        tmp_e, nullptr, nullptr, nullptr, eW2T[l], eb2[l], h_e, NE);
    // attention logits + segment softmax over dst
    logit_kernel<<<(NE * 32 + 255) / 256, 256, 0, stream>>>(h_e, aW[l], ab[l], wbuf, NE);
    init_layer_kernel<<<(NN * HD + 255) / 256, 256, 0, stream>>>(agg, mx, den, NN);
    attn_max_kernel<<<(NE + 255) / 256, 256, 0, stream>>>(wbuf, dst, mx, NE);
    attn_exp_kernel<<<(NE + 255) / 256, 256, 0, stream>>>(wbuf, dst, mx, den, NE);
    agg_scatter_kernel<<<((size_t)NE * HD + 255) / 256, 256, 0, stream>>>(
        h_e, wbuf, den, dst, agg, NE);
    // node MLP: hidden = relu([h_n|agg] @ nW1 + nb1); h_n = hidden @ nW2 + nb2
    gemm_wmma_kernel<MODE_NODECAT, 256, true><<<gN, 256, 0, stream>>>(
        agg, h_n, nullptr, nullptr, nW1T[l], nb1[l], tmp_n, NN);
    gemm_wmma_kernel<MODE_PLAIN, 128, false><<<gN, 256, 0, stream>>>(
        tmp_n, nullptr, nullptr, nullptr, nW2T[l], nb2[l], h_n, NN);
  }

  // ---- head ----
  head_kernel<<<(NN + 7) / 8, 256, 0, stream>>>(h_n, hW0, hb0, hW1, hb1, hW2, hb2,
                                                out, NN);
}
